// Kpcnn_49160195670356
// MI455X (gfx1250) — matrix operand order repaired and checked by reading, not compiled, not run on hardware
//
#include <hip/hip_runtime.h>
#include <hip/hip_bf16.h>

// ---------------------------------------------------------------------------
// Kernel-prediction CNN for MI455X (gfx1250, wave32, WMMA).
//
// Conv-as-GEMM per output pixel:  M = 16 batch (wave tile), N = cout tiles of
// 16, K = 9 taps x 32 padded channels -> 9 chunks of v_wmma_f32_16x16x32_f16.
// Activations LDS-resident in [b][pixel][32ch] f16 layout so each A-fragment
// K-chunk is two ds_load_b128 per lane (matches the 16x32 f16 A VGPR layout).
// Weights (B fragments) built once per layer, kept in VGPRs.
//
// Inner loop: all 9 A-fragments staged first, then 18 back-to-back WMMAs
// (two independent accumulator chains), px unrolled so edge-column OOB
// selects fold at compile time and adjacent pixels software-pipeline.
// ---------------------------------------------------------------------------

typedef _Float16 f16;
typedef _Float16 v16h __attribute__((ext_vector_type(16)));
typedef _Float16 v8h  __attribute__((ext_vector_type(8)));
typedef float    v8f  __attribute__((ext_vector_type(8)));

#define WAVES   4          // waves per workgroup
#define BT      16         // batch tile per wave (WMMA M)
#define WGB     (WAVES*BT) // 64 patches per workgroup
#define NPIX    25
#define CPAD    32         // channel padding (K per tap)
#define NUM_MID 6
#define NUM_W   6

// LDS (dynamic):
//  act   [2][WAVES][BT][NPIX][CPAD] f16   = 204800 B   (ping-pong activations)
//  hbuf  [WAVES][BT][NPIX][8]       f32   =  51200 B   (pre-softmax logits)
//  cols  [WAVES][BT][20]            f32   =   5120 B   (postconv colors, 18 used)
//  zrow  [WAVES][CPAD]              f16   =    256 B   (zero row for OOB taps)
#define ACT_HALVES  (2*WAVES*BT*NPIX*CPAD)
#define HBUF_FLOATS (WAVES*BT*NPIX*8)
#define COL_FLOATS  (WAVES*BT*20)
#define LDS_BYTES   (ACT_HALVES*2 + HBUF_FLOATS*4 + COL_FLOATS*4 + WAVES*CPAD*2)

template<int CIN, int COUT, bool RELU, bool TO_H>
__device__ __forceinline__ void conv_layer(
    f16* __restrict__ act, float* __restrict__ hbuf, const f16* __restrict__ zrow,
    int srcbuf, int dstbuf, int wv, int lane,
    const float* __restrict__ W, const float* __restrict__ bias)
{
    constexpr int NT = (COUT > 16) ? 2 : 1;
    const int co0   = lane & 15;   // N index (output channel within tile)
    const int half  = lane >> 4;   // which K-half of A / which B K-range
    const int kbase = half * 16;

    // ---- build B fragments (weights, f32->f16), kept in VGPRs ----
    // B is 32x16 (KxN): lanes 0-15 hold K{0..7,16..23}, lanes 16-31 hold K{8..15,24..31},
    // N = lane&15, 2 halves per VGPR low->high.
    v16h Bf[NT][9];
    #pragma unroll
    for (int t = 0; t < 9; ++t) {
        #pragma unroll
        for (int n = 0; n < NT; ++n) {
            const int co = co0 + n * 16;
            v16h bf;
            #pragma unroll
            for (int j = 0; j < 16; ++j) {
                const int ci = (j < 8) ? (half * 8 + j) : (16 + half * 8 + (j - 8));
                float w = 0.0f;
                if (co < COUT && ci < CIN) w = W[(co * CIN + ci) * 9 + t];
                bf[j] = (f16)w;
            }
            Bf[n][t] = bf;
        }
    }
    float bv[NT];
    #pragma unroll
    for (int n = 0; n < NT; ++n) {
        const int co = co0 + n * 16;
        bv[n] = (co < COUT) ? bias[co] : 0.0f;
    }

    const int b = lane & 15; // A-matrix row (batch) held by this lane
    const f16* srcb = act + ((size_t)(srcbuf * WAVES + wv) * BT + b) * (NPIX * CPAD);

    for (int py = 0; py < 5; ++py) {
        #pragma unroll
        for (int px = 0; px < 5; ++px) {
            const int p = py * 5 + px;

            v8f acc[NT];
            #pragma unroll
            for (int n = 0; n < NT; ++n) {
                #pragma unroll
                for (int r = 0; r < 8; ++r) acc[n][r] = bv[n];
            }

            // ---- stage all 9 A fragments (2x ds_load_b128 each) ----
            // A 16x32 f16 layout: lanes 0-15 K{0..7,16..23}, lanes 16-31
            // K{8..15,24..31} -> two contiguous 16B LDS loads per lane.
            union AFrag { v16h v; v8h h[2]; };
            AFrag A[9];
            #pragma unroll
            for (int t = 0; t < 9; ++t) {
                const int dy = t / 3 - 1;
                const int dx = t % 3 - 1;
                const int xx = px + dx;         // compile-time constant
                const f16* ap;
                if (xx < 0 || xx > 4) {
                    ap = zrow;                  // folded at compile time
                } else {
                    const int yy = py + dy;     // runtime
                    ap = ((unsigned)yy < 5u) ? (srcb + (yy * 5 + xx) * CPAD) : zrow;
                }
                A[t].h[0] = *(const v8h*)(ap + half * 8);
                A[t].h[1] = *(const v8h*)(ap + 16 + half * 8);
            }

            // ---- 9 (or 18) back-to-back WMMAs, two independent chains ----
            #pragma unroll
            for (int t = 0; t < 9; ++t) {
                acc[0] = __builtin_amdgcn_wmma_f32_16x16x32_f16(
                    false, A[t].v, false, Bf[0][t], (short)0, acc[0], false, false);
                if (NT == 2)
                    acc[1] = __builtin_amdgcn_wmma_f32_16x16x32_f16(
                        false, A[t].v, false, Bf[1][t], (short)0, acc[1], false, false);
            }

            // ---- writeback: C/D layout lane->N=co, vgpr r -> M=r(+8*half) ----
            if (TO_H) {
                if (co0 < COUT) {
                    #pragma unroll
                    for (int r = 0; r < 8; ++r) {
                        const int bb = r + half * 8;
                        hbuf[((wv * BT + bb) * NPIX + p) * 8 + co0] = acc[0][r];
                    }
                }
            } else {
                #pragma unroll
                for (int n = 0; n < NT; ++n) {
                    const int co = co0 + n * 16;
                    if (co < COUT) {
                        #pragma unroll
                        for (int r = 0; r < 8; ++r) {
                            const int bb = r + half * 8;
                            float v = acc[n][r];
                            if (RELU) v = fmaxf(v, 0.0f);
                            act[(((size_t)(dstbuf * WAVES + wv) * BT + bb) * NPIX + p) * CPAD + co] = (f16)v;
                        }
                    }
                }
            }
        }
    }
}

__global__ __launch_bounds__(WAVES * 32, 1)
void kpcnn_kernel(const float* __restrict__ input,
                  const float* __restrict__ w0,    const float* __restrict__ b0,
                  const float* __restrict__ wmid,  const float* __restrict__ bmid,
                  const float* __restrict__ wlast, const float* __restrict__ blast,
                  const float* __restrict__ wpost, const float* __restrict__ bpost,
                  float* __restrict__ out)
{
    extern __shared__ __align__(16) char smem[];
    f16*   act  = (f16*)smem;
    float* hbuf = (float*)(smem + ACT_HALVES * 2);
    float* cols = (float*)(smem + ACT_HALVES * 2 + HBUF_FLOATS * 4);
    f16*   zbase = (f16*)(smem + ACT_HALVES * 2 + HBUF_FLOATS * 4 + COL_FLOATS * 4);

    const int tid  = threadIdx.x;
    const int wv   = tid >> 5;
    const int lane = tid & 31;
    const int g0   = blockIdx.x * WGB + wv * BT;  // first patch of this wave
    f16* zrow = zbase + wv * CPAD;

    // ---- zero this wave's act slices + zero row (no cross-wave deps) ----
    uint4 z4; z4.x = z4.y = z4.z = z4.w = 0u;
    #pragma unroll
    for (int buf = 0; buf < 2; ++buf) {
        uint4* z = (uint4*)(act + (size_t)(buf * WAVES + wv) * BT * NPIX * CPAD);
        for (int i = lane; i < BT * NPIX * CPAD / 8; i += 32) z[i] = z4;
    }
    if (lane < CPAD / 8) ((uint4*)zrow)[lane] = z4;

    // ---- stage input (NCHW f32 -> [b][pix][ch] f16) into buffer 0 ----
    for (int i = lane; i < BT * 8 * NPIX; i += 32) {
        const int b = i / (8 * NPIX);
        const int r = i % (8 * NPIX);
        const int c = r / NPIX;
        const int p = r % NPIX;
        const float v = input[((size_t)(g0 + b) * 8 + c) * NPIX + p];
        act[(((size_t)wv * BT + b) * NPIX + p) * CPAD + c] = (f16)v;
    }

    // ---- conv stack (all WMMA) ----
    conv_layer<8, 25, true, false>(act, hbuf, zrow, 0, 1, wv, lane, w0, b0);
    int src = 1;
    for (int i = 0; i < NUM_MID; ++i) {
        conv_layer<25, 25, true, false>(act, hbuf, zrow, src, 1 - src, wv, lane,
                                        wmid + (size_t)i * 25 * 25 * 9, bmid + i * 25);
        src = 1 - src;
    }
    conv_layer<25, NUM_W, false, true>(act, hbuf, zrow, src, 1 - src, wv, lane, wlast, blast);

    // ---- postconv colors: 5x5 VALID on first 3 input channels -> [b][3][6] ----
    for (int i = lane; i < BT * 18; i += 32) {
        const int b = i / 18;
        const int o = i % 18;
        float s = bpost[o];
        const float* ip = input + (size_t)(g0 + b) * 8 * NPIX;
        const float* wp = wpost + (size_t)o * 3 * NPIX;
        #pragma unroll
        for (int c = 0; c < 3; ++c)
            #pragma unroll
            for (int p = 0; p < NPIX; ++p)
                s = fmaf(ip[c * NPIX + p], wp[c * NPIX + p], s);
        cols[(wv * BT + b) * 20 + o] = s;
    }

    // ---- softmax over NUM_W + weighted color blend ----
    for (int i = lane; i < BT * NPIX; i += 32) {
        const int b = i / NPIX;
        const int p = i % NPIX;
        const float* h = hbuf + ((size_t)(wv * BT + b) * NPIX + p) * 8;
        float m = h[0];
        #pragma unroll
        for (int w = 1; w < NUM_W; ++w) m = fmaxf(m, h[w]);
        float e[NUM_W], s = 0.0f;
        #pragma unroll
        for (int w = 0; w < NUM_W; ++w) { e[w] = __expf(h[w] - m); s += e[w]; }
        const float inv = 1.0f / s;
        const float* cb = cols + (size_t)(wv * BT + b) * 20;
        #pragma unroll
        for (int c = 0; c < 3; ++c) {
            float a = 0.0f;
            #pragma unroll
            for (int w = 0; w < NUM_W; ++w) a = fmaf(cb[c * NUM_W + w], e[w] * inv, a);
            out[((size_t)(g0 + b) * 3 + c) * NPIX + p] = a;
        }
    }
}

extern "C" void kernel_launch(void* const* d_in, const int* in_sizes, int n_in,
                              void* d_out, int out_size, void* d_ws, size_t ws_size,
                              hipStream_t stream) {
    (void)in_sizes; (void)n_in; (void)d_ws; (void)ws_size; (void)out_size;
    const float* input = (const float*)d_in[0];
    const float* w0    = (const float*)d_in[1];
    const float* b0    = (const float*)d_in[2];
    const float* wmid  = (const float*)d_in[3];
    const float* bmid  = (const float*)d_in[4];
    const float* wlast = (const float*)d_in[5];
    const float* blast = (const float*)d_in[6];
    const float* wpost = (const float*)d_in[7];
    const float* bpost = (const float*)d_in[8];
    float* out = (float*)d_out;

    const int B = 32768;
    dim3 grid(B / WGB);
    dim3 block(WAVES * 32);
    kpcnn_kernel<<<grid, block, LDS_BYTES, stream>>>(
        input, w0, b0, wmid, bmid, wlast, blast, wpost, bpost, out);
}
